// MobiusGRU_12000138625356
// MI455X (gfx1250) — matrix operand order repaired and checked by reading, not compiled, not run on hardware
//
#include <hip/hip_runtime.h>
#include <math.h>

// ---------------------------------------------------------------------------
// MobiusGRU for MI455X (gfx1250).  T=256, B=64, D=H=512.
// Bulk-precompute input-side mobius_matvec per layer (big bf16 WMMA GEMM +
// rowwise normalize); sequential scan = three small WMMA GEMMs per step with
// TDM (tensor_load_to_lds) double-buffered weight tiles + LDS-reduction
// Mobius elementwise kernels.
// ---------------------------------------------------------------------------

typedef __attribute__((ext_vector_type(16))) __bf16        v16bf;
typedef __attribute__((ext_vector_type(8)))  float         v8f;
typedef __attribute__((ext_vector_type(4)))  int           v4i;
typedef __attribute__((ext_vector_type(8)))  int           v8i;
typedef __attribute__((ext_vector_type(4)))  unsigned int  v4u;

#define EPSF 1e-15f
#define ACLIP 0.99999f   // 1 - 1e-5

#if defined(__HIP_DEVICE_COMPILE__) && \
    __has_builtin(__builtin_amdgcn_tensor_load_to_lds) && \
    __has_builtin(__builtin_amdgcn_s_wait_tensorcnt)
#define HAVE_TDM 1
#else
#define HAVE_TDM 0
#endif

__device__ __forceinline__ float clip_at(float x) {
  return fminf(fmaxf(x, -ACLIP), ACLIP);
}
__device__ __forceinline__ float sigmoidf_(float x) {
  return 1.0f / (1.0f + expf(-x));
}

// 256-thread block tree reduction through LDS
__device__ __forceinline__ float block_sum(float v, float* sbuf) {
  const int tid = threadIdx.x;
  sbuf[tid] = v;
  __syncthreads();
  #pragma unroll
  for (int s = 128; s > 0; s >>= 1) {
    if (tid < s) sbuf[tid] += sbuf[tid + s];
    __syncthreads();
  }
  float r = sbuf[0];
  __syncthreads();
  return r;
}

__device__ __forceinline__ void mobius_add2(float x0, float x1, float y0, float y1,
                                            float* sbuf, float& o0, float& o1) {
  float x2 = block_sum(x0 * x0 + x1 * x1, sbuf);
  float y2 = block_sum(y0 * y0 + y1 * y1, sbuf);
  float xy = block_sum(x0 * y0 + x1 * y1, sbuf);
  float den = fmaxf(1.0f + 2.0f * xy + x2 * y2, EPSF);
  float ca = 1.0f + 2.0f * xy + y2;
  float cb = 1.0f - x2;
  o0 = (ca * x0 + cb * y0) / den;
  o1 = (ca * x1 + cb * y1) / den;
}

// one_rnn_transform tail + logmap0 (pre-sigmoid) for one gate.
__device__ __forceinline__ void gate_compute(float m0, float m1, float u0, float u1,
                                             float bb0, float bb1, float hn, float ath,
                                             float* sbuf, float& o0, float& o1) {
  float mn2 = block_sum(m0 * m0 + m1 * m1, sbuf);
  float mraw = sqrtf(mn2);
  float mn = fmaxf(mraw, EPSF);
  float sc = tanhf(mn / hn * ath) / mn;
  if (mraw <= EPSF) sc = 0.0f;
  float a0 = m0 * sc, a1 = m1 * sc;          // mobius_matvec(W, hx)
  float c0, c1;
  mobius_add2(a0, a1, u0, u1, sbuf, c0, c1); // + Ux
  float e0, e1;
  mobius_add2(c0, c1, bb0, bb1, sbuf, e0, e1); // + bias
  float en2 = block_sum(e0 * e0 + e1 * e1, sbuf);
  float en = fmaxf(sqrtf(en2), EPSF);
  float ls = atanhf(clip_at(en)) / en;       // logmap0
  o0 = ls * e0;
  o1 = ls * e1;
}

// ---------------------------------------------------------------------------
// TDM: async load of a [64 rows x 32 k] bf16 tile (row stride = ldb elements)
// from global into LDS at byte offset lds_off. D# per CDNA5 ISA ch.8.
// 6-arg builtin form (clang-23 / therock-10.0 headers).
// ---------------------------------------------------------------------------
__device__ __forceinline__ void tdm_load_tile(unsigned lds_off, const void* gsrc,
                                              int ldb, int nB) {
#if HAVE_TDM
  unsigned long long ga = (unsigned long long)(uintptr_t)gsrc;
  v4u g0;
  g0[0] = 1u;                                       // count=1, user descriptor
  g0[1] = lds_off;                                  // lds_addr
  g0[2] = (unsigned)ga;                             // global_addr[31:0]
  g0[3] = (unsigned)((ga >> 32) & 0x1FFFFFFu) | (2u << 30);  // addr[56:32] | type=2
  v8i g1;
  g1[0] = 0x00010000;                               // wg_mask=0, data_size=1 (2B)
  g1[1] = (int)(((unsigned)ldb & 0xFFFFu) << 16);   // tensor_dim0 lo16 (len = ldb)
  g1[2] = (int)((((unsigned)ldb >> 16) & 0xFFFFu) |
                (((unsigned)nB & 0xFFFFu) << 16));  // dim0 hi | tensor_dim1 lo
  g1[3] = (int)((((unsigned)nB >> 16) & 0xFFFFu) |
                (32u << 16));                       // dim1 hi | tile_dim0 = 32
  g1[4] = 64;                                       // tile_dim1 = 64, tile_dim2 = 0
  g1[5] = ldb;                                      // tensor_dim0_stride lo32
  g1[6] = 0;                                        // stride hi | dim1_stride lo
  g1[7] = 0;
  v4i z4 = {0, 0, 0, 0};
  v8i z8 = {0, 0, 0, 0, 0, 0, 0, 0};
  __builtin_amdgcn_tensor_load_to_lds(g0, g1, z4, z4, z8, 0);
#else
  (void)lds_off; (void)gsrc; (void)ldb; (void)nB;
#endif
}

// ---------------------------------------------------------------------------
// small utility kernels
// ---------------------------------------------------------------------------
__global__ void cvt_f32_bf16(const float* __restrict__ s, __bf16* __restrict__ d, int n) {
  int i = blockIdx.x * blockDim.x + threadIdx.x;
  if (i < n) d[i] = (__bf16)s[i];
}

__global__ void init_h_kernel(const float* __restrict__ h0, float* __restrict__ H,
                              __bf16* __restrict__ HBF, int n) {
  int i = blockIdx.x * blockDim.x + threadIdx.x;
  if (i < n) {
    float v = h0[i];
    H[i] = v;
    HBF[i] = (__bf16)v;
  }
}

__global__ void copy_f32(const float* __restrict__ s, float* __restrict__ d, int n) {
  int i = blockIdx.x * blockDim.x + threadIdx.x;
  if (i < n) d[i] = s[i];
}

// ---------------------------------------------------------------------------
// WMMA GEMM:  C[M,N](f32) = A[M,K](bf16 row-major) @ B[N,K](bf16 row-major)^T
// Block tile (16*WAVES) x 64.  B tiles double-buffered in LDS via TDM
// (tensor_load_to_lds + s_wait_tensorcnt), fallback: cooperative b128 loads.
// Fragment packing per CDNA5 ISA 7.12.2 (16-bit A 16x32 / B 32x16).
// ---------------------------------------------------------------------------
template <int WAVES>
__global__ __launch_bounds__(WAVES * 32)
void gemm_bf16_nt(const __bf16* __restrict__ A, int lda,
                  const __bf16* __restrict__ B, int ldb, int nB,
                  float* __restrict__ C, int ldc, int K) {
  __shared__ __bf16 lds_b[2][64 * 32];  // two 4KB tiles: 64 cols x 32 k
  const int tid = threadIdx.x;
  const int wave = tid >> 5;
  const int lane = tid & 31;
  const int lh = lane >> 4;  // K-half select
  const int ll = lane & 15;  // row (A) / col (B,C)
  const int row0 = blockIdx.x * (16 * WAVES) + wave * 16;
  const int col0 = blockIdx.y * 64;

  v8f acc[4];
  const v8f vzero = {0.f, 0.f, 0.f, 0.f, 0.f, 0.f, 0.f, 0.f};
  #pragma unroll
  for (int j = 0; j < 4; ++j) acc[j] = vzero;

  const __bf16* Arow = A + (size_t)(row0 + ll) * lda;
  int cur = 0;

#if HAVE_TDM
  if (tid < 32)  // one TDM issue per workgroup (wave 0)
    tdm_load_tile((unsigned)(uintptr_t)&lds_b[0][0], B + (size_t)col0 * ldb, ldb, nB);
#endif

  for (int k0 = 0; k0 < K; k0 += 32) {
#if HAVE_TDM
    if (tid < 32) __builtin_amdgcn_s_wait_tensorcnt(0);
    __syncthreads();  // current buffer ready for all waves
    if ((k0 + 32 < K) && tid < 32)
      tdm_load_tile((unsigned)(uintptr_t)&lds_b[cur ^ 1][0],
                    B + (size_t)col0 * ldb + (k0 + 32), ldb, nB);
#else
    __syncthreads();
    for (int c = tid; c < 64 * 4; c += WAVES * 32) {
      int n = c >> 2;
      int kk = (c & 3) * 8;
      *(v4i*)&lds_b[0][n * 32 + kk] = *(const v4i*)&B[(size_t)(col0 + n) * ldb + k0 + kk];
      if (k0 + 32 < K)
        __builtin_prefetch(&B[(size_t)(col0 + n) * ldb + k0 + 32 + kk], 0, 1);
    }
    __syncthreads();
#endif

    // A fragment: lane row = ll; elems 0..7 = K(lh*8..+7), 8..15 = K(16+lh*8..+7)
    union Fr { v16bf v; v4i i2[2]; } af, bfr[4];
    af.i2[0] = *(const v4i*)&Arow[k0 + lh * 8];
    af.i2[1] = *(const v4i*)&Arow[k0 + 16 + lh * 8];
    // gather all four B fragments first so the ds_loads batch under one wait
    const __bf16* lb = lds_b[cur];
    #pragma unroll
    for (int j = 0; j < 4; ++j) {
      const __bf16* bp = &lb[(j * 16 + ll) * 32 + lh * 8];
      bfr[j].i2[0] = *(const v4i*)&bp[0];
      bfr[j].i2[1] = *(const v4i*)&bp[16];
    }
    // back-to-back XDL issues
    #pragma unroll
    for (int j = 0; j < 4; ++j)
      acc[j] = __builtin_amdgcn_wmma_f32_16x16x32_bf16(
          false, af.v, false, bfr[j].v, (short)0, acc[j], false, false);

#if HAVE_TDM
    __syncthreads();  // all waves done reading before buffer is refilled
    cur ^= 1;
#endif
  }

  // C/D layout: VGPR i, lanes 0-15 -> M=i, lanes 16-31 -> M=8+i; N=ll
  #pragma unroll
  for (int j = 0; j < 4; ++j)
    #pragma unroll
    for (int i = 0; i < 8; ++i)
      C[(size_t)(row0 + i + lh * 8) * ldc + col0 + j * 16 + ll] = acc[j][i];
}

// ---------------------------------------------------------------------------
// Bulk mobius_matvec normalization (in-place on MX rows of 512 per gate).
// ---------------------------------------------------------------------------
__global__ __launch_bounds__(256)
void ux_post(float* __restrict__ MX, const float* __restrict__ X) {
  __shared__ float sbuf[256];
  const int idx = blockIdx.x;
  const int tb = idx / 3;
  const int g = idx % 3;
  const float* x = X + (size_t)tb * 512;
  float* m = MX + (size_t)tb * 1536 + g * 512;
  const int d0 = threadIdx.x * 2;

  float x0 = x[d0], x1 = x[d0 + 1];
  float m0 = m[d0], m1 = m[d0 + 1];
  float xn2 = block_sum(x0 * x0 + x1 * x1, sbuf);
  float mn2 = block_sum(m0 * m0 + m1 * m1, sbuf);
  float xn = fmaxf(sqrtf(xn2), EPSF);
  float mraw = sqrtf(mn2);
  float mn = fmaxf(mraw, EPSF);
  float sc = tanhf(mn / xn * atanhf(clip_at(xn))) / mn;
  if (mraw <= EPSF) sc = 0.0f;
  m[d0] = m0 * sc;
  m[d0 + 1] = m1 * sc;
}

// ---------------------------------------------------------------------------
// Scan phase 1: z_t, r_t, rh_t.  One block per batch row.
// CZR: [B,1024] = [Whr@h | Whz@h].  UXt: [B,1536] gates (r,h,z).
// ---------------------------------------------------------------------------
__global__ __launch_bounds__(256)
void elem_zr(const float* __restrict__ Hst, const float* __restrict__ CZR,
             const float* __restrict__ UXt, const float* __restrict__ bias,
             float* __restrict__ ZT, float* __restrict__ RHF, __bf16* __restrict__ RHBF) {
  __shared__ float sbuf[256];
  const int b = blockIdx.x;
  const int d0 = threadIdx.x * 2;
  const float* hx = Hst + (size_t)b * 512;
  float h0v = hx[d0], h1v = hx[d0 + 1];
  float hn2 = block_sum(h0v * h0v + h1v * h1v, sbuf);
  float hn = fmaxf(sqrtf(hn2), EPSF);
  float ath = atanhf(clip_at(hn));

  // z gate (weight rows 1024..1535, Ux offset 1024, bias row 2)
  {
    const float* mz = CZR + (size_t)b * 1024 + 512;
    const float* uz = UXt + (size_t)b * 1536 + 1024;
    float z0, z1;
    gate_compute(mz[d0], mz[d0 + 1], uz[d0], uz[d0 + 1],
                 bias[1024 + d0], bias[1024 + d0 + 1], hn, ath, sbuf, z0, z1);
    ZT[(size_t)b * 512 + d0] = sigmoidf_(z0);
    ZT[(size_t)b * 512 + d0 + 1] = sigmoidf_(z1);
  }
  // r gate (weight rows 0..511, Ux offset 0, bias row 0)
  float r0, r1;
  {
    const float* mr = CZR + (size_t)b * 1024;
    const float* ur = UXt + (size_t)b * 1536;
    gate_compute(mr[d0], mr[d0 + 1], ur[d0], ur[d0 + 1],
                 bias[d0], bias[d0 + 1], hn, ath, sbuf, r0, r1);
    r0 = sigmoidf_(r0);
    r1 = sigmoidf_(r1);
  }
  // rh_t = mobius_pointwise_mul(r_t, hx)
  float w0 = r0 * h0v, w1 = r1 * h1v;
  float wn2 = block_sum(w0 * w0 + w1 * w1, sbuf);
  float wraw = sqrtf(wn2);
  float wn = fmaxf(wraw, EPSF);
  float sc = tanhf(wn / hn * ath) / wn;
  if (wraw <= EPSF) sc = 0.0f;
  float rh0 = w0 * sc, rh1 = w1 * sc;
  RHF[(size_t)b * 512 + d0] = rh0;
  RHF[(size_t)b * 512 + d0 + 1] = rh1;
  RHBF[(size_t)b * 512 + d0] = (__bf16)rh0;
  RHBF[(size_t)b * 512 + d0 + 1] = (__bf16)rh1;
}

// ---------------------------------------------------------------------------
// Scan phase 2: h_tilde, delta, h_new; writes H/HBF/out.
// ---------------------------------------------------------------------------
__global__ __launch_bounds__(256)
void elem_h(float* __restrict__ Hst, const float* __restrict__ CH,
            const float* __restrict__ UXt, const float* __restrict__ bias,
            const float* __restrict__ ZT, const float* __restrict__ RHF,
            __bf16* __restrict__ HBF, float* __restrict__ OUTF, __bf16* OUTBF) {
  __shared__ float sbuf[256];
  const int b = blockIdx.x;
  const int d0 = threadIdx.x * 2;
  float* hx = Hst + (size_t)b * 512;
  float h0v = hx[d0], h1v = hx[d0 + 1];

  // x_norm for mobius_matvec(W_hh, rh_t) is ||rh||
  const float* rh = RHF + (size_t)b * 512;
  float rr0 = rh[d0], rr1 = rh[d0 + 1];
  float rn2 = block_sum(rr0 * rr0 + rr1 * rr1, sbuf);
  float rn = fmaxf(sqrtf(rn2), EPSF);
  float atr = atanhf(clip_at(rn));

  const float* m = CH + (size_t)b * 512;
  float m0 = m[d0], m1 = m[d0 + 1];
  float mn2 = block_sum(m0 * m0 + m1 * m1, sbuf);
  float mraw = sqrtf(mn2);
  float mn = fmaxf(mraw, EPSF);
  float sc = tanhf(mn / rn * atr) / mn;
  if (mraw <= EPSF) sc = 0.0f;
  float a0 = m0 * sc, a1 = m1 * sc;

  const float* u = UXt + (size_t)b * 1536 + 512;   // gate h
  float c0, c1;
  mobius_add2(a0, a1, u[d0], u[d0 + 1], sbuf, c0, c1);
  float e0, e1;                                     // h_tilde
  mobius_add2(c0, c1, bias[512 + d0], bias[512 + d0 + 1], sbuf, e0, e1);

  // delta = mobius_add(-hx, h_tilde)
  float x2 = block_sum(h0v * h0v + h1v * h1v, sbuf);
  float y2 = block_sum(e0 * e0 + e1 * e1, sbuf);
  float xy = block_sum(-(h0v * e0 + h1v * e1), sbuf);
  float den = fmaxf(1.0f + 2.0f * xy + x2 * y2, EPSF);
  float dl0 = ((1.0f + 2.0f * xy + y2) * (-h0v) + (1.0f - x2) * e0) / den;
  float dl1 = ((1.0f + 2.0f * xy + y2) * (-h1v) + (1.0f - x2) * e1) / den;

  // zd = mobius_pointwise_mul(z_t, delta)
  float z0 = ZT[(size_t)b * 512 + d0], z1 = ZT[(size_t)b * 512 + d0 + 1];
  float dn2 = block_sum(dl0 * dl0 + dl1 * dl1, sbuf);
  float dn = fmaxf(sqrtf(dn2), EPSF);
  float w0 = z0 * dl0, w1 = z1 * dl1;
  float wn2 = block_sum(w0 * w0 + w1 * w1, sbuf);
  float wraw = sqrtf(wn2);
  float wn = fmaxf(wraw, EPSF);
  float sc2 = tanhf(wn / dn * atanhf(clip_at(dn))) / wn;
  if (wraw <= EPSF) sc2 = 0.0f;
  float zd0 = w0 * sc2, zd1 = w1 * sc2;

  // h_new = mobius_add(hx, zd)
  y2 = block_sum(zd0 * zd0 + zd1 * zd1, sbuf);
  xy = block_sum(h0v * zd0 + h1v * zd1, sbuf);
  den = fmaxf(1.0f + 2.0f * xy + x2 * y2, EPSF);
  float n0 = ((1.0f + 2.0f * xy + y2) * h0v + (1.0f - x2) * zd0) / den;
  float n1 = ((1.0f + 2.0f * xy + y2) * h1v + (1.0f - x2) * zd1) / den;

  hx[d0] = n0;
  hx[d0 + 1] = n1;
  HBF[(size_t)b * 512 + d0] = (__bf16)n0;
  HBF[(size_t)b * 512 + d0 + 1] = (__bf16)n1;
  OUTF[(size_t)b * 512 + d0] = n0;
  OUTF[(size_t)b * 512 + d0 + 1] = n1;
  if (OUTBF) {
    OUTBF[(size_t)b * 512 + d0] = (__bf16)n0;
    OUTBF[(size_t)b * 512 + d0 + 1] = (__bf16)n1;
  }
}

// ---------------------------------------------------------------------------
// host driver
// ---------------------------------------------------------------------------
extern "C" void kernel_launch(void* const* d_in, const int* in_sizes, int n_in,
                              void* d_out, int out_size, void* d_ws, size_t ws_size,
                              hipStream_t stream) {
  (void)in_sizes; (void)n_in; (void)out_size; (void)ws_size;
  const int T = 256, B = 64, Hd = 512, G = 1536;

  const float* in_x   = (const float*)d_in[0];
  const float* in_h0  = (const float*)d_in[1];
  const float* w_ih1  = (const float*)d_in[2];
  const float* w_ih2  = (const float*)d_in[3];
  const float* w_hh1  = (const float*)d_in[4];
  const float* w_hh2  = (const float*)d_in[5];
  const float* bias1  = (const float*)d_in[6];
  const float* bias2  = (const float*)d_in[7];
  float* out = (float*)d_out;

  // workspace carve-out
  char* w = (char*)d_ws;
  size_t off = 0;
  auto carve = [&](size_t bytes) {
    void* p = w + off;
    off += (bytes + 255) & ~(size_t)255;
    return p;
  };
  __bf16* bWih1 = (__bf16*)carve((size_t)G * Hd * 2);
  __bf16* bWhh1 = (__bf16*)carve((size_t)G * Hd * 2);
  __bf16* bWih2 = (__bf16*)carve((size_t)G * Hd * 2);
  __bf16* bWhh2 = (__bf16*)carve((size_t)G * Hd * 2);
  __bf16* XBF   = (__bf16*)carve((size_t)T * B * Hd * 2);  // layer input bf16; reused as out1 bf16
  float*  MX    = (float*)carve((size_t)T * B * G * 4);    // input-side GEMM / Ux (per layer)
  float*  OUT1  = (float*)carve((size_t)T * B * Hd * 4);   // layer-1 outputs f32
  float*  Hbuf  = (float*)carve((size_t)B * Hd * 4);
  __bf16* HBF   = (__bf16*)carve((size_t)B * Hd * 2);
  float*  CZR   = (float*)carve((size_t)B * 1024 * 4);
  float*  CH    = (float*)carve((size_t)B * Hd * 4);
  float*  ZT    = (float*)carve((size_t)B * Hd * 4);
  float*  RHF   = (float*)carve((size_t)B * Hd * 4);
  __bf16* RHBF  = (__bf16*)carve((size_t)B * Hd * 2);

  // weight + input conversion to bf16
  {
    int nw = G * Hd;
    cvt_f32_bf16<<<(nw + 255) / 256, 256, 0, stream>>>(w_ih1, bWih1, nw);
    cvt_f32_bf16<<<(nw + 255) / 256, 256, 0, stream>>>(w_hh1, bWhh1, nw);
    cvt_f32_bf16<<<(nw + 255) / 256, 256, 0, stream>>>(w_ih2, bWih2, nw);
    cvt_f32_bf16<<<(nw + 255) / 256, 256, 0, stream>>>(w_hh2, bWhh2, nw);
    int nx = T * B * Hd;
    cvt_f32_bf16<<<(nx + 255) / 256, 256, 0, stream>>>(in_x, XBF, nx);
  }

  const size_t outOff = (size_t)T * B * Hd;  // start of ht in d_out

  for (int layer = 0; layer < 2; ++layer) {
    const float*  xf32  = (layer == 0) ? in_x : OUT1;
    const __bf16* wihbf = (layer == 0) ? bWih1 : bWih2;
    const __bf16* whhbf = (layer == 0) ? bWhh1 : bWhh2;
    const float*  bias  = (layer == 0) ? bias1 : bias2;
    float*  outf  = (layer == 0) ? OUT1 : out;
    __bf16* outbf = (layer == 0) ? XBF : (__bf16*)nullptr;  // layer-2 input bf16

    // bulk input-side GEMM: MX[T*B, 1536] = Xbf @ Wih^T ; then mobius normalize
    gemm_bf16_nt<8><<<dim3(T * B / 128, G / 64), 256, 0, stream>>>(
        XBF, Hd, wihbf, Hd, G, MX, G, Hd);
    ux_post<<<T * B * 3, 256, 0, stream>>>(MX, xf32);

    init_h_kernel<<<(B * Hd + 255) / 256, 256, 0, stream>>>(
        in_h0 + (size_t)layer * B * Hd, Hbuf, HBF, B * Hd);

    for (int t = 0; t < T; ++t) {
      const float* UXt = MX + (size_t)t * B * G;
      // r half: W_hr rows [0,512)  -> CZR cols [0,512)
      gemm_bf16_nt<4><<<dim3(1, 8), 128, 0, stream>>>(
          HBF, Hd, whhbf, Hd, Hd, CZR, 1024, Hd);
      // z half: W_hz rows [1024,1536) -> CZR cols [512,1024)
      gemm_bf16_nt<4><<<dim3(1, 8), 128, 0, stream>>>(
          HBF, Hd, whhbf + (size_t)1024 * Hd, Hd, Hd, CZR + 512, 1024, Hd);
      elem_zr<<<B, 256, 0, stream>>>(Hbuf, CZR, UXt, bias, ZT, RHF, RHBF);
      // h_tilde GEMM: W_hh rows [512,1024) with rh_t
      gemm_bf16_nt<4><<<dim3(1, 8), 128, 0, stream>>>(
          RHBF, Hd, whhbf + (size_t)512 * Hd, Hd, Hd, CH, Hd, Hd);
      elem_h<<<B, 256, 0, stream>>>(
          Hbuf, CH, UXt, bias, ZT, RHF, HBF,
          outf + (size_t)t * B * Hd,
          outbf ? (outbf + (size_t)t * B * Hd) : (__bf16*)nullptr);
    }
    // ht[layer] = final hidden state
    copy_f32<<<(B * Hd + 255) / 256, 256, 0, stream>>>(
        Hbuf, out + outOff + (size_t)layer * B * Hd, B * Hd);
  }
}
